// KDAAttention_56753697849915
// MI455X (gfx1250) — matrix operand order repaired and checked by reading, not compiled, not run on hardware
//
#include <hip/hip_runtime.h>
#include <hip/hip_bf16.h>
#include <stdint.h>

// ---------------------------------------------------------------------------
// KDA gated delta-rule attention for MI455X (gfx1250, wave32, WMMA)
//   B=4 T=1024 DIM=2048 H=16 D=128 K=4 P=2048
// GEMMs: v_wmma_f32_16x16x32_bf16; B K-tiles double-buffered via
//        global_load_async_to_lds_b128 (+ s_wait_asynccnt), A double-buffered
//        in registers, all 8 B fragments batch-loaded from LDS per K-step.
// Scan:  128x128 fp32 state resident in LDS, one workgroup per (b,h).
// ---------------------------------------------------------------------------

#define Bb   4
#define Tt   1024
#define DIMd 2048
#define Hh   16
#define Dd   128
#define Pp   2048
#define BT   (Bb * Tt)          // 4096 rows in every GEMM

typedef __bf16 bf16_t;
typedef __bf16 v16bf __attribute__((ext_vector_type(16)));
typedef float  v8f   __attribute__((ext_vector_type(8)));

union Frag16 { v16bf v; uint4 u[2]; };

__device__ __forceinline__ float silu_f(float x)    { return x / (1.f + __expf(-x)); }
__device__ __forceinline__ float sigmoid_f(float x) { return 1.f / (1.f + __expf(-x)); }

// CDNA5 async global->LDS copy (ASYNCcnt-tracked). lds_addr = LDS byte offset
// (low 32 bits of a generic LDS pointer), gsrc = 64-bit global address.
__device__ __forceinline__ void async_b128(unsigned lds_addr, const void* gsrc) {
    asm volatile("global_load_async_to_lds_b128 %0, %1, off"
                 :: "v"(lds_addr), "v"(gsrc) : "memory");
}
__device__ __forceinline__ void wait_async2() {
    asm volatile("s_wait_asynccnt 0x2" ::: "memory");
}
__device__ __forceinline__ void wait_async0() {
    asm volatile("s_wait_asynccnt 0x0" ::: "memory");
}

// ---------------------------------------------------------------------------
// fp32 -> bf16 (vectorized x4)
// ---------------------------------------------------------------------------
__global__ __launch_bounds__(256)
void k_f32_to_bf16(const float* __restrict__ in, bf16_t* __restrict__ out, int n4) {
    int i = (blockIdx.x * 256 + threadIdx.x);
    if (i < n4) {
        float4 f = ((const float4*)in)[i];
        bf16_t* o = out + (size_t)i * 4;
        o[0] = (bf16_t)f.x; o[1] = (bf16_t)f.y; o[2] = (bf16_t)f.z; o[3] = (bf16_t)f.w;
    }
}

// ---------------------------------------------------------------------------
// Repack weight W[Kd x N] (row-major fp32) into WMMA B-fragment layout, bf16.
// Tile = 32(K) x 16(N). Per tile, 32 lanes x 16 bf16 contiguous:
//   lane<16 : N=lane,    K = kt*32 + 0..15
//   lane>=16: N=lane-16, K = kt*32 + 16..31
// ---------------------------------------------------------------------------
__global__ __launch_bounds__(256)
void k_pack_w(const float* __restrict__ W, bf16_t* __restrict__ Pk, int Kd, int N) {
    int gid = blockIdx.x * 256 + threadIdx.x;     // one thread per (tile, lane)
    int ntiles = N >> 4;
    int total  = (Kd >> 5) * ntiles * 32;
    if (gid >= total) return;
    int lane = gid & 31;
    int tile = gid >> 5;
    int nt   = tile % ntiles;
    int kt   = tile / ntiles;
    int n     = nt * 16 + (lane & 15);
    int kbase = kt * 32 + ((lane < 16) ? 0 : 16);
    bf16_t* dst = Pk + (size_t)tile * 512 + (size_t)lane * 16;
#pragma unroll
    for (int j = 0; j < 16; ++j)
        dst[j] = (bf16_t)W[(size_t)(kbase + j) * N + n];
}

// ---------------------------------------------------------------------------
// C[M x N] (fp32) = A[M x Kd] (bf16 row-major) * Bp (packed bf16)
// Block: 256 threads = 8 waves; block tile 128 x (NT*16).
// NT==8: B K-tiles (8KB) staged into LDS with double-buffered async copies,
//        prefetched one K-step ahead; A double-buffered in registers; all 8
//        B fragments batch-loaded (one DS clause) before the 8 WMMAs.
// NT==1: (N=16 beta projection) direct global B loads, no LDS.
// ---------------------------------------------------------------------------
template <int NT>
__global__ __launch_bounds__(256)
void k_gemm_bf16(const bf16_t* __restrict__ A, const bf16_t* __restrict__ Bp,
                 float* __restrict__ C, int M, int N, int Kd) {
    extern __shared__ char lds[];                 // NT==8: 2 x 8KB B buffers
    const int tid   = threadIdx.x;
    const int lane  = tid & 31;
    const int wave  = tid >> 5;
    const int mbase = blockIdx.x * 128 + wave * 16;
    const int ntiles_total = N >> 4;
    const int ntile0 = blockIdx.y * NT;

    v8f acc[NT];
#pragma unroll
    for (int i = 0; i < NT; ++i)
#pragma unroll
        for (int j = 0; j < 8; ++j) acc[i][j] = 0.f;

    const int mrow = mbase + (lane & 15);
    const int koff = (lane < 16) ? 0 : 16;
    const bf16_t* arow = A + (size_t)mrow * Kd + koff;

    if constexpr (NT == 8) {
        const int KT = Kd >> 5;
        const size_t btile_bytes = (size_t)ntiles_total * 1024;  // bytes per K-step
        const char* bsrc = (const char*)(Bp + (size_t)ntile0 * 512);
        const int tb = tid * 32;                            // this thread's 32B slice
        const unsigned lbase = (unsigned)(size_t)(&lds[0]);
        const unsigned d0 = lbase + tb;
        const unsigned d1 = lbase + 8192 + tb;

        // prefetch K-step 0 into LDS buffer 0, A fragment 0 into registers
        async_b128(d0, bsrc + tb);
        async_b128(d0 + 16, bsrc + tb + 16);
        Frag16 af;
        af.u[0] = *(const uint4*)(arow);
        af.u[1] = *(const uint4*)(arow + 8);

        for (int kt = 0; kt < KT; ++kt) {
            const int cur = kt & 1;
            if (kt + 1 < KT) {                              // prefetch next B K-step
                const char* s = bsrc + (size_t)(kt + 1) * btile_bytes + tb;
                unsigned d = cur ? d0 : d1;
                async_b128(d, s);
                async_b128(d + 16, s + 16);
                wait_async2();                              // current tile landed
            } else {
                wait_async0();
            }
            __syncthreads();

            // batch-load all 8 B fragments (single DS clause, cascaded waits)
            const char* buf = lds + (cur ? 8192 : 0) + lane * 32;
            Frag16 bfg[8];
#pragma unroll
            for (int nt = 0; nt < 8; ++nt) {
                bfg[nt].u[0] = *(const uint4*)(buf + nt * 1024);
                bfg[nt].u[1] = *(const uint4*)(buf + nt * 1024 + 16);
            }
            // prefetch next A fragment (branchless: clamp to last K-step)
            const int ktn = (kt + 1 < KT) ? (kt + 1) : kt;
            Frag16 afn;
            afn.u[0] = *(const uint4*)(arow + ktn * 32);
            afn.u[1] = *(const uint4*)(arow + ktn * 32 + 8);

#pragma unroll
            for (int nt = 0; nt < 8; ++nt)
                acc[nt] = __builtin_amdgcn_wmma_f32_16x16x32_bf16(
                    false, af.v, false, bfg[nt].v, (short)0, acc[nt], false, false);
            af = afn;
            __syncthreads();                                // buffer WAR fence
        }
    } else {
        const bf16_t* bcol = Bp + ((size_t)ntile0 * 512) + (size_t)lane * 16;
        const size_t bkstep = (size_t)ntiles_total * 512;
#pragma unroll 2
        for (int k0 = 0; k0 < Kd; k0 += 32) {
            Frag16 af;
            af.u[0] = *(const uint4*)(arow + k0);
            af.u[1] = *(const uint4*)(arow + k0 + 8);
            const bf16_t* bb = bcol + (size_t)(k0 >> 5) * bkstep;
#pragma unroll
            for (int nt = 0; nt < NT; ++nt) {
                Frag16 bfg;
                bfg.u[0] = *(const uint4*)(bb + (size_t)nt * 512);
                bfg.u[1] = *(const uint4*)(bb + (size_t)nt * 512 + 8);
                acc[nt] = __builtin_amdgcn_wmma_f32_16x16x32_bf16(
                    false, af.v, false, bfg.v, (short)0, acc[nt], false, false);
            }
        }
    }

    // C/D layout: VGPR r -> M = r (lanes 0-15) / r+8 (lanes 16-31); N = lane&15
    const int r_add = (lane < 16) ? 0 : 8;
    const int col0  = ntile0 * 16 + (lane & 15);
#pragma unroll
    for (int nt = 0; nt < NT; ++nt) {
        int col = col0 + nt * 16;
#pragma unroll
        for (int r = 0; r < 8; ++r)
            C[(size_t)(mbase + r + r_add) * N + col] = acc[nt][r];
    }
}

// ---------------------------------------------------------------------------
// Causal depthwise conv (K=4) + SiLU + per-head L2 norm for q/k, plus the
// KDA gate g = -exp(A_log)*softplus(gpre+dt_bias) and beta = sigmoid(bpre).
// One block per (b,t), 256 threads, P=2048 channels.
// ---------------------------------------------------------------------------
__device__ void conv_silu(const float* __restrict__ pre, const float* __restrict__ w,
                          float* __restrict__ ybuf, size_t base, int t, int tid) {
#pragma unroll
    for (int i = 0; i < 8; ++i) {
        int c = tid + i * 256;
        float acc = 0.f;
#pragma unroll
        for (int j = 0; j < 4; ++j) {
            int tt = t - 3 + j;
            if (tt >= 0)
                acc += pre[(long)base + (long)(tt - t) * Pp + c] * w[c * 4 + j];
        }
        ybuf[c] = silu_f(acc);
    }
}

__global__ __launch_bounds__(256)
void k_conv_gate(const float* __restrict__ qpre, const float* __restrict__ kpre,
                 const float* __restrict__ vpre,
                 const float* __restrict__ conv_q, const float* __restrict__ conv_k,
                 const float* __restrict__ conv_v,
                 const float* __restrict__ gpre, const float* __restrict__ bpre,
                 const float* __restrict__ A_log, const float* __restrict__ dt_bias,
                 float* __restrict__ qo, float* __restrict__ ko, float* __restrict__ vo,
                 float* __restrict__ go, float* __restrict__ betao) {
    __shared__ float ybuf[Pp];
    __shared__ float hn[Hh];
    const int bt  = blockIdx.x;
    const int t   = bt & (Tt - 1);
    const int tid = threadIdx.x;
    const size_t base = (size_t)bt * Pp;

    // ---- q: conv + silu + head-wise l2norm * D^-0.5 ----
    conv_silu(qpre, conv_q, ybuf, base, t, tid);
    __syncthreads();
    if (tid < Hh) {
        float s = 0.f;
        for (int d = 0; d < Dd; ++d) { float y = ybuf[tid * Dd + d]; s += y * y; }
        hn[tid] = rsqrtf(s + 1e-6f) * 0.08838834764831845f;   // * 1/sqrt(128)
    }
    __syncthreads();
#pragma unroll
    for (int i = 0; i < 8; ++i) { int c = tid + i * 256; qo[base + c] = ybuf[c] * hn[c >> 7]; }
    __syncthreads();

    // ---- k: conv + silu + head-wise l2norm ----
    conv_silu(kpre, conv_k, ybuf, base, t, tid);
    __syncthreads();
    if (tid < Hh) {
        float s = 0.f;
        for (int d = 0; d < Dd; ++d) { float y = ybuf[tid * Dd + d]; s += y * y; }
        hn[tid] = rsqrtf(s + 1e-6f);
    }
    __syncthreads();
#pragma unroll
    for (int i = 0; i < 8; ++i) { int c = tid + i * 256; ko[base + c] = ybuf[c] * hn[c >> 7]; }
    __syncthreads();

    // ---- v: conv + silu ----
    conv_silu(vpre, conv_v, ybuf, base, t, tid);
    __syncthreads();
#pragma unroll
    for (int i = 0; i < 8; ++i) { int c = tid + i * 256; vo[base + c] = ybuf[c]; }

    // ---- gate g and beta ----
#pragma unroll
    for (int i = 0; i < 8; ++i) {
        int c = tid + i * 256, h = c >> 7;
        float x  = gpre[base + c] + dt_bias[c];
        float sp = (x > 20.f) ? x : log1pf(__expf(x));
        go[base + c] = -__expf(A_log[h]) * sp;
    }
    if (tid < Hh)
        betao[(size_t)bt * Hh + tid] = sigmoid_f(bpre[(size_t)bt * Hh + tid]);
}

// ---------------------------------------------------------------------------
// Recurrent KDA scan. One block per (b,h); S[128][128] fp32 lives in LDS.
// 256 threads: thread owns column v = tid&127, row-half = tid>>7.
// Dynamic LDS: 64KB state + staging + partial sums = 69632 bytes.
// ---------------------------------------------------------------------------
__global__ __launch_bounds__(256)
void k_scan(const float* __restrict__ q, const float* __restrict__ k,
            const float* __restrict__ v, const float* __restrict__ g,
            const float* __restrict__ beta, float* __restrict__ o) {
    extern __shared__ char smem[];
    float* S    = (float*)smem;                 // 128*128 fp32
    float* kk   = (float*)(smem + 65536);       // 128
    float* qq   = kk + 128;
    float* vv   = qq + 128;
    float* eg   = vv + 128;
    float* part = eg + 128;                     // 256

    const int bh   = blockIdx.x;                // b*H + h
    const int tid  = threadIdx.x;
    const int vcol = tid & 127;
    const int half = tid >> 7;

    for (int i = tid; i < Dd * Dd; i += 256) S[i] = 0.f;
    __syncthreads();

    const size_t rowstride = (size_t)Hh * Dd;   // step per t in [B,T,H,D]
    size_t idx  = ((size_t)(bh >> 4) * Tt * Hh + (bh & 15)) * Dd;
    size_t bidx = (size_t)(bh >> 4) * Tt * Hh + (bh & 15);

    for (int t = 0; t < Tt; ++t) {
        if (tid < Dd) {
            kk[tid] = k[idx + tid];
            qq[tid] = q[idx + tid];
            vv[tid] = v[idx + tid];
            eg[tid] = __expf(g[idx + tid]);
        }
        const float bb = beta[bidx];
        __syncthreads();

        // decay + partial column dot  k . (S * exp(g))
        float acc = 0.f;
#pragma unroll 4
        for (int i = 0; i < 64; ++i) {
            int kr = half * 64 + i;
            float s = S[kr * Dd + vcol] * eg[kr];
            S[kr * Dd + vcol] = s;
            acc += kk[kr] * s;
        }
        part[tid] = acc;
        __syncthreads();
        const float err = vv[vcol] - (part[vcol] + part[vcol + 128]);
        const float upd = bb * err;
        __syncthreads();                        // part[] reused below

        // rank-1 update + partial output dot  q . S
        float acc2 = 0.f;
#pragma unroll 4
        for (int i = 0; i < 64; ++i) {
            int kr = half * 64 + i;
            float s = S[kr * Dd + vcol] + kk[kr] * upd;
            S[kr * Dd + vcol] = s;
            acc2 += qq[kr] * s;
        }
        part[tid] = acc2;
        __syncthreads();
        if (tid < Dd) o[idx + vcol] = part[vcol] + part[vcol + 128];
        __syncthreads();

        idx  += rowstride;
        bidx += Hh;
    }
}

// ---------------------------------------------------------------------------
// Gated RMSNorm (sigmoid gate) -> bf16, ready for the WMMA output projection.
// ---------------------------------------------------------------------------
__global__ __launch_bounds__(256)
void k_norm_gate(const float* __restrict__ o, const float* __restrict__ gate,
                 const float* __restrict__ w, bf16_t* __restrict__ out) {
    __shared__ float ybuf[Pp];
    __shared__ float red[Hh];
    const int bt  = blockIdx.x;
    const int tid = threadIdx.x;
    const size_t base = (size_t)bt * Pp;
#pragma unroll
    for (int i = 0; i < 8; ++i) { int c = tid + i * 256; ybuf[c] = o[base + c]; }
    __syncthreads();
    if (tid < Hh) {
        float s = 0.f;
        for (int d = 0; d < Dd; ++d) { float y = ybuf[tid * Dd + d]; s += y * y; }
        red[tid] = rsqrtf(s * (1.f / Dd) + 1e-5f);
    }
    __syncthreads();
#pragma unroll
    for (int i = 0; i < 8; ++i) {
        int c = tid + i * 256, h = c >> 7, d = c & 127;
        float val = ybuf[c] * red[h] * w[d] * sigmoid_f(gate[base + c]);
        out[base + c] = (bf16_t)val;
    }
}

// ---------------------------------------------------------------------------
// Host driver
// ---------------------------------------------------------------------------
extern "C" void kernel_launch(void* const* d_in, const int* in_sizes, int n_in,
                              void* d_out, int out_size, void* d_ws, size_t ws_size,
                              hipStream_t stream) {
    (void)in_sizes; (void)n_in; (void)out_size; (void)ws_size;

    const float* x      = (const float*)d_in[0];
    const float* Wq     = (const float*)d_in[1];
    const float* Wk     = (const float*)d_in[2];
    const float* Wv     = (const float*)d_in[3];
    const float* conv_q = (const float*)d_in[4];
    const float* conv_k = (const float*)d_in[5];
    const float* conv_v = (const float*)d_in[6];
    const float* A_log  = (const float*)d_in[7];
    const float* dtb    = (const float*)d_in[8];
    const float* Wfa    = (const float*)d_in[9];
    const float* Wfb    = (const float*)d_in[10];
    const float* Wb     = (const float*)d_in[11];
    const float* Wga    = (const float*)d_in[12];
    const float* Wgb    = (const float*)d_in[13];
    const float* onw    = (const float*)d_in[14];
    const float* Wo     = (const float*)d_in[15];
    float* out = (float*)d_out;

    char* ws = (char*)d_ws;
    size_t off = 0;
    auto alloc = [&](size_t bytes) -> char* {
        char* p = ws + off;
        off += (bytes + 255) & ~(size_t)255;
        return p;
    };

    bf16_t* xb    = (bf16_t*)alloc((size_t)BT * DIMd * 2);    // also reused for onorm
    bf16_t* Wq_p  = (bf16_t*)alloc((size_t)DIMd * Pp * 2);
    bf16_t* Wk_p  = (bf16_t*)alloc((size_t)DIMd * Pp * 2);
    bf16_t* Wv_p  = (bf16_t*)alloc((size_t)DIMd * Pp * 2);
    bf16_t* Wo_p  = (bf16_t*)alloc((size_t)Pp * DIMd * 2);
    bf16_t* Wfa_p = (bf16_t*)alloc((size_t)DIMd * Dd * 2);
    bf16_t* Wfb_p = (bf16_t*)alloc((size_t)Dd * Pp * 2);
    bf16_t* Wga_p = (bf16_t*)alloc((size_t)DIMd * Dd * 2);
    bf16_t* Wgb_p = (bf16_t*)alloc((size_t)Dd * Pp * 2);
    bf16_t* Wb_p  = (bf16_t*)alloc((size_t)DIMd * Hh * 2);
    float* qpre   = (float*)alloc((size_t)BT * Pp * 4);       // also reused for o (scan out)
    float* kpre   = (float*)alloc((size_t)BT * Pp * 4);
    float* vpre   = (float*)alloc((size_t)BT * Pp * 4);
    float* f1     = (float*)alloc((size_t)BT * Dd * 4);
    float* g1     = (float*)alloc((size_t)BT * Dd * 4);
    bf16_t* f1b   = (bf16_t*)alloc((size_t)BT * Dd * 2);
    bf16_t* g1b   = (bf16_t*)alloc((size_t)BT * Dd * 2);
    float* gpre   = (float*)alloc((size_t)BT * Pp * 4);
    float* gateb  = (float*)alloc((size_t)BT * Pp * 4);
    float* bpre   = (float*)alloc((size_t)BT * Hh * 4);
    float* qn     = (float*)alloc((size_t)BT * Pp * 4);
    float* kn     = (float*)alloc((size_t)BT * Pp * 4);
    float* vn     = (float*)alloc((size_t)BT * Pp * 4);
    float* gbuf   = (float*)alloc((size_t)BT * Pp * 4);
    float* betab  = (float*)alloc((size_t)BT * Hh * 4);
    float* obuf   = qpre;                                     // alias (qpre dead after conv)
    bf16_t* onorm = xb;                                       // alias (x dead after GEMMs)

    // 1) convert x to bf16
    {
        int n4 = BT * DIMd / 4;
        k_f32_to_bf16<<<(n4 + 255) / 256, 256, 0, stream>>>(x, xb, n4);
    }
    // 2) pack all weights into WMMA B-fragment layout
    auto pack = [&](const float* W, bf16_t* Pk, int Kd, int N) {
        int total = (Kd / 32) * (N / 16) * 32;
        k_pack_w<<<(total + 255) / 256, 256, 0, stream>>>(W, Pk, Kd, N);
    };
    pack(Wq, Wq_p, DIMd, Pp);   pack(Wk, Wk_p, DIMd, Pp);
    pack(Wv, Wv_p, DIMd, Pp);   pack(Wo, Wo_p, Pp, DIMd);
    pack(Wfa, Wfa_p, DIMd, Dd); pack(Wfb, Wfb_p, Dd, Pp);
    pack(Wga, Wga_p, DIMd, Dd); pack(Wgb, Wgb_p, Dd, Pp);
    pack(Wb, Wb_p, DIMd, Hh);

    // N multiple of 128 -> NT=8 (async LDS staging, 16KB dyn LDS);
    // N==16 -> NT=1 (direct loads).
    auto gemm = [&](const bf16_t* A, const bf16_t* Bp, float* C, int M, int N, int Kd) {
        if (N >= 128) {
            dim3 grid(M / 128, N / 128);
            k_gemm_bf16<8><<<grid, 256, 16384, stream>>>(A, Bp, C, M, N, Kd);
        } else {
            dim3 grid(M / 128, 1);
            k_gemm_bf16<1><<<grid, 256, 0, stream>>>(A, Bp, C, M, N, Kd);
        }
    };

    // 3) big projections + low-rank first stages
    gemm(xb, Wq_p, qpre, BT, Pp, DIMd);
    gemm(xb, Wk_p, kpre, BT, Pp, DIMd);
    gemm(xb, Wv_p, vpre, BT, Pp, DIMd);
    gemm(xb, Wfa_p, f1, BT, Dd, DIMd);
    gemm(xb, Wga_p, g1, BT, Dd, DIMd);
    gemm(xb, Wb_p, bpre, BT, Hh, DIMd);

    // 4) low-rank second stages (convert intermediates to bf16 first)
    {
        int n4 = BT * Dd / 4;
        k_f32_to_bf16<<<(n4 + 255) / 256, 256, 0, stream>>>(f1, f1b, n4);
        k_f32_to_bf16<<<(n4 + 255) / 256, 256, 0, stream>>>(g1, g1b, n4);
    }
    gemm(f1b, Wfb_p, gpre, BT, Pp, Dd);
    gemm(g1b, Wgb_p, gateb, BT, Pp, Dd);

    // 5) conv + silu + l2norm + gates
    k_conv_gate<<<BT, 256, 0, stream>>>(qpre, kpre, vpre, conv_q, conv_k, conv_v,
                                        gpre, bpre, A_log, dtb,
                                        qn, kn, vn, gbuf, betab);

    // 6) recurrent KDA scan (64 blocks, 69632 B dynamic LDS each)
    k_scan<<<Bb * Hh, 256, 69632, stream>>>(qn, kn, vn, gbuf, betab, obuf);

    // 7) gated RMSNorm -> bf16
    k_norm_gate<<<BT, 256, 0, stream>>>(obuf, gateb, onw, onorm);

    // 8) output projection
    gemm(onorm, Wo_p, out, BT, DIMd, Pp);
}
